// DCNv2_31207232373051
// MI455X (gfx1250) — matrix operand order repaired and checked by reading, not compiled, not run on hardware
//
#include <hip/hip_runtime.h>
#include <hip/hip_bf16.h>

typedef __attribute__((ext_vector_type(16))) __bf16 v16bf;
typedef __attribute__((ext_vector_type(8)))  float  v8f;

constexpr int Hc = 112, Wc = 112, Bc = 8, CIN = 64, COUT = 128;
constexpr int KTAP = 9, KF = CIN * KTAP;       // 576 contraction length
constexpr int LDSROW = 584;                     // padded LDS row stride (halves), 16B aligned
constexpr int TW = Wc / 16;                     // 7 pixel-tiles per row
constexpr int HW = Hc * Wc;
constexpr int KSTEPS = KF / 32;                 // 18 WMMA k-steps

// fp32 -> bf16 round-to-nearest-even
__device__ __forceinline__ unsigned short f2bf(float f) {
    unsigned u = __float_as_uint(f);
    u += 0x7FFFu + ((u >> 16) & 1u);
    return (unsigned short)(u >> 16);
}

// A fragment (16x32 bf16, M = lane&15): K slices [k0+8h, +8) and [k0+16+8h, +8)
__device__ __forceinline__ v16bf load_fragA(const unsigned short* row, int k0, int hl) {
    union { uint4 q[2]; v16bf v; } u;
    u.q[0] = *(const uint4*)(row + k0 + 8 * hl);
    u.q[1] = *(const uint4*)(row + k0 + 16 + 8 * hl);
    return u.v;
}

// B fragment (32x16 bf16, N = lane&15): K slices [k0+8h, +8) and [k0+16+8h, +8)
__device__ __forceinline__ v16bf load_fragB(const unsigned short* row, int k0, int hl) {
    union { uint4 q[2]; v16bf v; } u;
    const unsigned short* p = row + k0 + 8 * hl;
    u.q[0] = *(const uint4*)(p);
    u.q[1] = *(const uint4*)(p + 16);
    return u.v;
}

// ---------------- kernel 1: repack weights to bf16, K-major (k = tap*64 + cin) ----
__global__ void __launch_bounds__(128) prep_weights(
    const float* __restrict__ w_conv, const float* __restrict__ w_off,
    const float* __restrict__ w_mask, unsigned short* __restrict__ wB,
    unsigned short* __restrict__ wOM)
{
    int idx = blockIdx.x * blockDim.x + threadIdx.x;
    if (idx < COUT * KF) {
        int o = idx / KF, r = idx % KF;
        int tap = r / CIN, c = r % CIN;
        wB[idx] = f2bf(w_conv[(o * CIN + c) * KTAP + tap]);
    }
    if (idx < 32 * KF) {                       // rows 0..17 offs, 18..26 mask, 27..31 zero
        int o = idx / KF, r = idx % KF;
        int tap = r / CIN, c = r % CIN;
        float v = 0.f;
        if (o < 18)      v = w_off [(o * CIN + c) * KTAP + tap];
        else if (o < 27) v = w_mask[((o - 18) * CIN + c) * KTAP + tap];
        wOM[idx] = f2bf(v);
    }
}

// ---------------- kernel 2: fused DCNv2 tile kernel ------------------------------
// stage 1: undeformed im2col -> LDS          (all 256 threads)
// stage 2: offset/mask WMMA  -> LDS offm     (waves 0-1; waves 2-7 prefetch weights)
// stage 3: deformed modulated im2col -> LDS  (all 256 threads)
// stage 4: main implicit GEMM via WMMA       (8 waves x 16 couts)
__global__ void __launch_bounds__(256) dcn_fused(
    const float* __restrict__ x, const unsigned short* __restrict__ wB,
    const unsigned short* __restrict__ wOM, float* __restrict__ out)
{
    __shared__ alignas(16) unsigned short samp[16 * LDSROW];   // 18,688 B
    __shared__ float offm[16][28];                             // [pixel][0..17=dy/dx, 18..26=mask]

    int tile = blockIdx.x;
    int b   = tile / (Hc * TW);
    int rem = tile % (Hc * TW);
    int h   = rem / TW;
    int w0  = (rem % TW) * 16;

    const float* xb = x + (size_t)b * CIN * HW;
    int lane = threadIdx.x & 31, wave = threadIdx.x >> 5;
    int hl = lane >> 4, n = lane & 15;

    // ---- stage 1: plain im2col (bf16) for the offset/mask conv ----
    for (int u = threadIdx.x; u < 16 * KTAP * 4; u += blockDim.x) {
        int pair = u >> 2, chunk = u & 3;                      // 4 chunks of 16 channels
        int p = pair / KTAP, tap = pair % KTAP;
        int yy = h + tap / 3 - 1;
        int xx = w0 + p + tap % 3 - 1;
        bool valid = (yy >= 0) & (yy < Hc) & (xx >= 0) & (xx < Wc);
        const float* src = xb + (size_t)(chunk * 16) * HW + yy * Wc + xx;
        unsigned short* dst = &samp[p * LDSROW + tap * CIN + chunk * 16];
        #pragma unroll
        for (int c = 0; c < 16; ++c)
            dst[c] = f2bf(valid ? src[(size_t)c * HW] : 0.f);
    }
    __syncthreads();

    // ---- stage 2: offset/mask GEMM on waves 0-1; weight prefetch on waves 2-7 ----
    if (wave < 2) {
        const unsigned short* arow = &samp[n * LDSROW];
        const unsigned short* brow = wOM + (size_t)(wave * 16 + n) * KF;
        v8f acc = {0.f, 0.f, 0.f, 0.f, 0.f, 0.f, 0.f, 0.f};
        for (int s = 0; s < KSTEPS; ++s) {
            v16bf a  = load_fragA(arow, s * 32, hl);
            v16bf bm = load_fragB(brow, s * 32, hl);
            acc = __builtin_amdgcn_wmma_f32_16x16x32_bf16(false, a, false, bm,
                                                          (short)0, acc, false, false);
        }
        int co = wave * 16 + n;
        if (co < 27) {
            #pragma unroll
            for (int j = 0; j < 8; ++j) {                      // D: M = j + 8*(lane/16)
                float v = acc[j];
                offm[j + 8 * hl][co] = (co < 18) ? v : 2.f / (1.f + __expf(-v));
            }
        }
    } else {
        // warm L2/L1 with this wave's B rows for stage 4 (global_prefetch_b8)
        const unsigned short* brow = wB + (size_t)(wave * 16 + n) * KF;
        #pragma unroll
        for (int s = 0; s < KSTEPS; ++s)
            __builtin_prefetch(brow + s * 32, 0, 3);
    }
    __syncthreads();

    // ---- stage 3: deformed, mask-modulated im2col (overwrites samp) ----
    for (int u = threadIdx.x; u < 16 * KTAP * 4; u += blockDim.x) {
        int pair = u >> 2, chunk = u & 3;
        int p = pair / KTAP, tap = pair % KTAP;
        int w = w0 + p;
        float dy = offm[p][2 * tap];
        float dx = offm[p][2 * tap + 1];
        float m  = offm[p][18 + tap];
        float py = (float)(h + tap / 3 - 1) + dy;
        float px = (float)(w + tap % 3 - 1) + dx;
        float fy = floorf(py), fx = floorf(px);
        float wy = py - fy,    wx = px - fx;
        int y0 = (int)fy, x0 = (int)fx, y1 = y0 + 1, x1 = x0 + 1;
        bool vy0 = (y0 >= 0) & (y0 < Hc), vy1 = (y1 >= 0) & (y1 < Hc);
        bool vx0 = (x0 >= 0) & (x0 < Wc), vx1 = (x1 >= 0) & (x1 < Wc);
        int cy0 = min(max(y0, 0), Hc - 1), cy1 = min(max(y1, 0), Hc - 1);
        int cx0 = min(max(x0, 0), Wc - 1), cx1 = min(max(x1, 0), Wc - 1);
        int i00 = cy0 * Wc + cx0, i01 = cy0 * Wc + cx1;
        int i10 = cy1 * Wc + cx0, i11 = cy1 * Wc + cx1;
        float w00 = (vy0 & vx0) ? (1.f - wy) * (1.f - wx) * m : 0.f;
        float w01 = (vy0 & vx1) ? (1.f - wy) * wx         * m : 0.f;
        float w10 = (vy1 & vx0) ? wy * (1.f - wx)         * m : 0.f;
        float w11 = (vy1 & vx1) ? wy * wx                 * m : 0.f;
        const float* base = xb + (size_t)(chunk * 16) * HW;
        unsigned short* dst = &samp[p * LDSROW + tap * CIN + chunk * 16];
        #pragma unroll
        for (int c = 0; c < 16; ++c) {
            const float* pc = base + (size_t)c * HW;
            float v = w00 * pc[i00] + w01 * pc[i01] + w10 * pc[i10] + w11 * pc[i11];
            dst[c] = f2bf(v);
        }
    }
    __syncthreads();

    // ---- stage 4: main implicit GEMM, one 16-cout tile per wave ----
    {
        const unsigned short* arow = &samp[n * LDSROW];
        const unsigned short* brow = wB + (size_t)(wave * 16 + n) * KF;
        v8f acc = {0.f, 0.f, 0.f, 0.f, 0.f, 0.f, 0.f, 0.f};
        for (int s = 0; s < KSTEPS; ++s) {
            v16bf a  = load_fragA(arow, s * 32, hl);
            v16bf bm = load_fragB(brow, s * 32, hl);
            acc = __builtin_amdgcn_wmma_f32_16x16x32_bf16(false, a, false, bm,
                                                          (short)0, acc, false, false);
        }
        int co = wave * 16 + n;
        size_t obase = (((size_t)b * COUT + co) * Hc + h) * Wc + w0 + 8 * hl;
        *(float4*)(out + obase)     = make_float4(acc[0], acc[1], acc[2], acc[3]);
        *(float4*)(out + obase + 4) = make_float4(acc[4], acc[5], acc[6], acc[7]);
    }
}

// ---------------- launcher -------------------------------------------------------
extern "C" void kernel_launch(void* const* d_in, const int* in_sizes, int n_in,
                              void* d_out, int out_size, void* d_ws, size_t ws_size,
                              hipStream_t stream) {
    const float* x      = (const float*)d_in[0];
    const float* w_conv = (const float*)d_in[1];
    const float* w_off  = (const float*)d_in[2];
    const float* w_mask = (const float*)d_in[3];
    float* out = (float*)d_out;

    unsigned char* ws = (unsigned char*)d_ws;
    unsigned short* wB  = (unsigned short*)ws;                  // 128*576*2 = 147456 B
    unsigned short* wOM = (unsigned short*)(ws + 147456);       // 32*576*2  =  36864 B

    int nprep = COUT * KF;                                      // 73728
    prep_weights<<<(nprep + 127) / 128, 128, 0, stream>>>(w_conv, w_off, w_mask, wB, wOM);

    int tiles = Bc * Hc * TW;                                   // 6272
    dcn_fused<<<tiles, 256, 0, stream>>>(x, wB, wOM, out);
}
